// MultiHeadAttention_55018531062336
// MI455X (gfx1250) — matrix operand order repaired and checked by reading, not compiled, not run on hardware
//
#include <hip/hip_runtime.h>

typedef __bf16 bf16_t;
typedef __attribute__((ext_vector_type(4)))  bf16_t v4bf;
typedef __attribute__((ext_vector_type(8)))  bf16_t v8bf;
typedef __attribute__((ext_vector_type(16))) bf16_t v16bf;
typedef __attribute__((ext_vector_type(8)))  float  v8f;
typedef __attribute__((ext_vector_type(4)))  float  v4f;

#define EMB 1024
#define NH  16
#define HD  64
#define BATCH 4
#define SEQ 2048
#define MROWS (BATCH*SEQ)   // 8192

union U16 { v16bf v; v8bf h[2]; };

// native bf16 convert (lowers to HW cvt, round-to-nearest-even)
static __device__ __forceinline__ bf16_t f2bf(float f) { return (bf16_t)f; }

static __device__ __forceinline__ v8f zero8() {
    v8f z;
#pragma unroll
    for (int i = 0; i < 8; ++i) z[i] = 0.0f;
    return z;
}

static __device__ __forceinline__ v8f wmma_bf16(v16bf a, v16bf b, v8f c) {
    return __builtin_amdgcn_wmma_f32_16x16x32_bf16(false, a, false, b, (short)0, c, false, false);
}

// A-fragment (16x32 bf16, row-major source): lane row = lane%16, hv = lane/16.
// elements 0..7  <- base[hv*8 + 0..7]        (K = hv*8 + e)
// elements 8..15 <- base[16 + hv*8 + 0..7]   (K = 16 + hv*8 + (e-8))
static __device__ __forceinline__ v16bf load_a16(const bf16_t* base, int hv) {
    U16 u;
    u.h[0] = *(const v8bf*)(base + hv * 8);
    u.h[1] = *(const v8bf*)(base + 16 + hv * 8);
    return u.v;
}

// B-fragment (32x16 bf16): lane col = lane%16, elements 0..7 <- base[hv*8 + e], 8..15 <- base[16 + hv*8 + (e-8)]
static __device__ __forceinline__ v16bf load_b16(const bf16_t* base, int hv) {
    U16 u;
    u.h[0] = *(const v8bf*)(base + hv * 8);
    u.h[1] = *(const v8bf*)(base + 16 + hv * 8);
    return u.v;
}

// ---------------- Stage 0: activation fp32 -> bf16 (streaming) -------------
__global__ void __launch_bounds__(256) x_convert_kernel(const float* __restrict__ X,
                                                        bf16_t* __restrict__ Xb) {
    int idx = (blockIdx.x * 256 + threadIdx.x) * 4;
    v4f f = *(const v4f*)(X + idx);
    v4bf o;
#pragma unroll
    for (int i = 0; i < 4; ++i) o[i] = f2bf(f[i]);
    *(v4bf*)(Xb + idx) = o;
}

// ---------------- Stage 1: weight transpose + fp32->bf16 -------------------
// Wt[n*EMB + k] = bf16( W[k*EMB + n] )  -> B-fragments become contiguous loads
__global__ void __launch_bounds__(256) wt_convert_kernel(const float* __restrict__ W,
                                                         bf16_t* __restrict__ Wt) {
    int idx = blockIdx.x * 256 + threadIdx.x;     // idx = n*EMB + k
    int nn = idx >> 10;
    int kk = idx & (EMB - 1);
    Wt[idx] = f2bf(W[kk * EMB + nn]);
}

// ---------------- Stage 2/4: unified bf16 GEMM -----------------------------
// C[M,N] = A_bf16[M,K] * Wt_bf16[N,K]^T + bias ; M=8192, N=K=1024
// Block tile 128x128 (8 waves), wave tile 64x32 (4x2 frags -> 8 WMMA / K-step)
// mode 0: out bf16 [b][h][s][d]     (Q,K)
// mode 1: out bf16 [b][h][d][s]     (V transposed)
// mode 2: out fp32 row-major [M,N]  (final projection)
__global__ void __launch_bounds__(256) gemm_kernel(const bf16_t* __restrict__ A,
                                                   const bf16_t* __restrict__ Wt,
                                                   const float* __restrict__ bias,
                                                   void* __restrict__ outp,
                                                   int mode) {
    const int lane = threadIdx.x & 31;
    const int wave = threadIdx.x >> 5;
    const int n  = lane & 15;
    const int hv = lane >> 4;
    const int wn = wave & 3, wmi = wave >> 2;
    const int m0 = blockIdx.y * 128 + wmi * 64;
    const int n0 = blockIdx.x * 128 + wn * 32;

    v8f c[4][2];
#pragma unroll
    for (int mi = 0; mi < 4; ++mi) { c[mi][0] = zero8(); c[mi][1] = zero8(); }

    const bf16_t* ap[4];
#pragma unroll
    for (int mi = 0; mi < 4; ++mi) ap[mi] = A + (size_t)(m0 + mi * 16 + n) * EMB;
    const bf16_t* w0 = Wt + (size_t)(n0 + n) * EMB;
    const bf16_t* w1 = Wt + (size_t)(n0 + 16 + n) * EMB;

    for (int k0 = 0; k0 < EMB; k0 += 32) {
        v16bf b0 = load_b16(w0 + k0, hv);
        v16bf b1 = load_b16(w1 + k0, hv);
#pragma unroll
        for (int mi = 0; mi < 4; ++mi) {
            v16bf a = load_a16(ap[mi] + k0, hv);
            c[mi][0] = wmma_bf16(a, b0, c[mi][0]);
            c[mi][1] = wmma_bf16(a, b1, c[mi][1]);
        }
    }

    const float bia[2] = { bias[n0 + n], bias[n0 + 16 + n] };

    if (mode == 2) {
        float* O = (float*)outp;
#pragma unroll
        for (int mi = 0; mi < 4; ++mi)
#pragma unroll
            for (int ni = 0; ni < 2; ++ni) {
                int col = n0 + ni * 16 + n;
#pragma unroll
                for (int r = 0; r < 8; ++r) {
                    int rr = m0 + mi * 16 + r + hv * 8;
                    O[(size_t)rr * EMB + col] = c[mi][ni][r] + bia[ni];
                }
            }
    } else {
        bf16_t* O = (bf16_t*)outp;
#pragma unroll
        for (int mi = 0; mi < 4; ++mi)
#pragma unroll
            for (int ni = 0; ni < 2; ++ni) {
                int col = n0 + ni * 16 + n;
                int hh = col >> 6, dd = col & 63;
#pragma unroll
                for (int r = 0; r < 8; ++r) {
                    int rr = m0 + mi * 16 + r + hv * 8;      // b*SEQ + s
                    int bb = rr >> 11, s = rr & (SEQ - 1);
                    float val = c[mi][ni][r] + bia[ni];
                    size_t oidx = (mode == 0)
                        ? ((size_t)((bb * NH + hh) * SEQ + s) * HD + dd)
                        : ((size_t)((bb * NH + hh) * HD + dd) * SEQ + s);
                    O[oidx] = f2bf(val);
                }
            }
    }
}

// ---------------- Stage 3: flash attention ---------------------------------
// Per wave: 16 queries, stream 32 keys/iter. Computes S^T = K*Q^T so the
// softmaxed P fragment is a pure in-lane register remap into the PV A-frag.
__global__ void __launch_bounds__(256) attn_kernel(const bf16_t* __restrict__ Qm,
                                                   const bf16_t* __restrict__ Km,
                                                   const bf16_t* __restrict__ Vt,
                                                   bf16_t* __restrict__ ctx) {
    const int lane = threadIdx.x & 31;
    const int n  = lane & 15;
    const int hv = lane >> 4;
    const int wave = threadIdx.x >> 5;
    const int q0 = blockIdx.x * 128 + wave * 16;
    const int bh = blockIdx.y;
    const int b = bh >> 4, h = bh & 15;

    const bf16_t* Qp = Qm + (size_t)bh * SEQ * HD;
    const bf16_t* Kp = Km + (size_t)bh * SEQ * HD;
    const bf16_t* Vp = Vt + (size_t)bh * HD * SEQ;

    // Q^T B-fragments (64 x 16), loaded once: B[kk][n] = Q[q0+n][kk] contiguous
    v16bf qb0 = load_b16(Qp + (size_t)(q0 + n) * HD, hv);
    v16bf qb1 = load_b16(Qp + (size_t)(q0 + n) * HD + 32, hv);

    v8f o[4];
#pragma unroll
    for (int i = 0; i < 4; ++i) o[i] = zero8();

    float m_run = -3.0e38f;
    float l_run = 0.0f;
    const float scale = 0.125f;   // 1/sqrt(64)

    for (int j0 = 0; j0 < SEQ; j0 += 32) {
        // K A-fragments: rows = keys, contraction = d
        v16bf ka00 = load_a16(Kp + (size_t)(j0 + n) * HD, hv);
        v16bf ka01 = load_a16(Kp + (size_t)(j0 + n) * HD + 32, hv);
        v16bf ka10 = load_a16(Kp + (size_t)(j0 + 16 + n) * HD, hv);
        v16bf ka11 = load_a16(Kp + (size_t)(j0 + 16 + n) * HD + 32, hv);

        v8f s0 = zero8(), s1 = zero8();
        s0 = wmma_bf16(ka00, qb0, s0);  s0 = wmma_bf16(ka01, qb1, s0);
        s1 = wmma_bf16(ka10, qb0, s1);  s1 = wmma_bf16(ka11, qb1, s1);

        // lane holds query n, keys {hv*8+0..7} (s0) and {16+hv*8+0..7} (s1)
        float sc[16];
#pragma unroll
        for (int r = 0; r < 8; ++r) { sc[r] = s0[r] * scale; sc[8 + r] = s1[r] * scale; }

        float mt = sc[0];
#pragma unroll
        for (int e = 1; e < 16; ++e) mt = fmaxf(mt, sc[e]);
        mt = fmaxf(mt, __shfl_xor(mt, 16, 32));           // combine key halves
        float mnew = fmaxf(m_run, mt);
        float alpha = __expf(m_run - mnew);

        float sum = 0.0f;
        U16 pa;                                            // P A-fragment (bf16)
#pragma unroll
        for (int e = 0; e < 16; ++e) {
            float pe = __expf(sc[e] - mnew);
            sum += pe;
            pa.v[e] = f2bf(pe);     // element order already matches A-frag K order
        }
        sum += __shfl_xor(sum, 16, 32);
        l_run = l_run * alpha + sum;
        m_run = mnew;

        // redistribute alpha (indexed by query=lane%16) into C-frag row order
        float ar[8];
#pragma unroll
        for (int r = 0; r < 8; ++r) ar[r] = __shfl(alpha, hv * 8 + r, 32);

#pragma unroll
        for (int db = 0; db < 4; ++db) {
            v16bf vb = load_b16(Vp + (size_t)(db * 16 + n) * SEQ + j0, hv);
#pragma unroll
            for (int r = 0; r < 8; ++r) o[db][r] *= ar[r];
            o[db] = wmma_bf16(pa.v, vb, o[db]);
        }
    }

    float lr[8];
#pragma unroll
    for (int r = 0; r < 8; ++r) lr[r] = __shfl(l_run, hv * 8 + r, 32);

#pragma unroll
    for (int db = 0; db < 4; ++db) {
#pragma unroll
        for (int r = 0; r < 8; ++r) {
            int q = q0 + r + hv * 8;
            ctx[(size_t)(b * SEQ + q) * EMB + h * HD + db * 16 + n] = f2bf(o[db][r] / lr[r]);
        }
    }
}

// ---------------------------------------------------------------------------
extern "C" void kernel_launch(void* const* d_in, const int* in_sizes, int n_in,
                              void* d_out, int out_size, void* d_ws, size_t ws_size,
                              hipStream_t stream) {
    const float* q  = (const float*)d_in[0];
    const float* k  = (const float*)d_in[1];
    const float* v  = (const float*)d_in[2];
    const float* Wq = (const float*)d_in[3];
    const float* Wk = (const float*)d_in[4];
    const float* Wv = (const float*)d_in[5];
    const float* Wo = (const float*)d_in[6];
    const float* bq = (const float*)d_in[7];
    const float* bk = (const float*)d_in[8];
    const float* bv = (const float*)d_in[9];
    const float* bo = (const float*)d_in[10];

    const size_t WT_BYTES  = (size_t)EMB * EMB * sizeof(bf16_t);   // 2 MB
    const size_t BIG_BYTES = (size_t)MROWS * EMB * sizeof(bf16_t); // 16 MB

    char* ws = (char*)d_ws;
    bf16_t* WtQ = (bf16_t*)(ws);
    bf16_t* WtK = (bf16_t*)(ws + 1 * WT_BYTES);
    bf16_t* WtV = (bf16_t*)(ws + 2 * WT_BYTES);
    bf16_t* WtO = (bf16_t*)(ws + 3 * WT_BYTES);
    bf16_t* Xq  = (bf16_t*)(ws + 4 * WT_BYTES);                       // reused as ctx
    bf16_t* Xk  = (bf16_t*)(ws + 4 * WT_BYTES + 1 * BIG_BYTES);
    bf16_t* Xv  = (bf16_t*)(ws + 4 * WT_BYTES + 2 * BIG_BYTES);
    bf16_t* Qb  = (bf16_t*)(ws + 4 * WT_BYTES + 3 * BIG_BYTES);
    bf16_t* Kb  = (bf16_t*)(ws + 4 * WT_BYTES + 4 * BIG_BYTES);
    bf16_t* Vtb = (bf16_t*)(ws + 4 * WT_BYTES + 5 * BIG_BYTES);
    bf16_t* ctx = Xq;   // Xq is dead after the Q projection

    // Stage 0: activations fp32 -> bf16 (streaming, 4 elems/thread)
    const int xgrid = (MROWS * EMB) / (256 * 4);   // 8192 blocks
    x_convert_kernel<<<xgrid, 256, 0, stream>>>(q, Xq);
    x_convert_kernel<<<xgrid, 256, 0, stream>>>(k, Xk);
    x_convert_kernel<<<xgrid, 256, 0, stream>>>(v, Xv);

    // Stage 1: transpose + convert the four weight matrices
    wt_convert_kernel<<<(EMB * EMB) / 256, 256, 0, stream>>>(Wq, WtQ);
    wt_convert_kernel<<<(EMB * EMB) / 256, 256, 0, stream>>>(Wk, WtK);
    wt_convert_kernel<<<(EMB * EMB) / 256, 256, 0, stream>>>(Wv, WtV);
    wt_convert_kernel<<<(EMB * EMB) / 256, 256, 0, stream>>>(Wo, WtO);

    // Stage 2: QKV projections (block tile 128x128, wave tile 64x32)
    dim3 gg(EMB / 128, MROWS / 128);   // (8, 64)
    gemm_kernel<<<gg, 256, 0, stream>>>(Xq, WtQ, bq, Qb, 0);
    gemm_kernel<<<gg, 256, 0, stream>>>(Xk, WtK, bk, Kb, 0);
    gemm_kernel<<<gg, 256, 0, stream>>>(Xv, WtV, bv, Vtb, 1);

    // Stage 3: flash attention (16 queries/wave, 128/block; one (b,h) per blockIdx.y)
    attn_kernel<<<dim3(SEQ / 128, BATCH * NH), 256, 0, stream>>>(Qb, Kb, Vtb, ctx);

    // Stage 4: output projection -> fp32 d_out
    gemm_kernel<<<gg, 256, 0, stream>>>(ctx, WtO, bo, d_out, 2);
}
